// RSMLayer_57904749084766
// MI455X (gfx1250) — matrix operand order repaired and checked by reading, not compiled, not run on hardware
//
#include <hip/hip_runtime.h>
#include <hip/hip_bf16.h>
#include <stdint.h>

typedef __bf16 bf16;
typedef __attribute__((ext_vector_type(16))) __bf16 v16bf;
typedef __attribute__((ext_vector_type(8)))  __bf16 v8bf;
typedef __attribute__((ext_vector_type(8)))  float  v8f;

#define BSZ     64
#define SEQ     64
#define D_IN    2048
#define D_OUT   2048
#define MDIM    1024
#define TCELL   8192
#define KSEL    128
#define GAMMA_C 0.5f
#define EPS_C   0.5f

// ---------------- workspace layout (bytes) ----------------
#define WS_WB    ((size_t)0)                              // 8192*8192 bf16 = 128MB
#define WS_WA    (WS_WB   + (size_t)TCELL*TCELL*2)
#define WS_WD    (WS_WA   + (size_t)MDIM*D_IN*2)
#define WS_XA_BF (WS_WD   + (size_t)D_OUT*MDIM*2)         // whole x_a batch in bf16 (16MB)
#define WS_XBBF  (WS_XA_BF+ (size_t)SEQ*BSZ*D_IN*2)       // x_b bf16 (1MB)
#define WS_GBF   (WS_XBBF + (size_t)BSZ*TCELL*2)          // group-max bf16 (128KB)
#define WS_XB    (WS_GBF  + (size_t)BSZ*MDIM*2)           // x_b fp32 state
#define WS_PHI   (WS_XB   + (size_t)BSZ*TCELL*4)
#define WS_PSI   (WS_PHI  + (size_t)BSZ*TCELL*4)
#define WS_ZA    (WS_PSI  + (size_t)BSZ*TCELL*4)
#define WS_SIG   (WS_ZA   + (size_t)BSZ*MDIM*4)
#define WS_PART  (WS_SIG  + (size_t)BSZ*TCELL*4)
#define WS_SCAL  (WS_PART + 256)

// ---------------- d_out layout (floats) -------------------
#define O_OUT   ((size_t)0)
#define O_XB    ((size_t)SEQ*BSZ*D_OUT)
#define O_PHI   (O_XB  + (size_t)BSZ*TCELL)
#define O_PSI   (O_PHI + (size_t)BSZ*TCELL)
#define O_XBS   (O_PSI + (size_t)BSZ*TCELL)

// monotonic float<->uint order transform (for atomicMin on floats)
__device__ __forceinline__ unsigned orderbits(float f) {
    unsigned u = __float_as_uint(f);
    return (u & 0x80000000u) ? ~u : (u | 0x80000000u);
}
__device__ __forceinline__ float unorder(unsigned o) {
    unsigned u = (o & 0x80000000u) ? (o ^ 0x80000000u) : ~o;
    return __uint_as_float(u);
}
// fp32 -> bf16 round-to-nearest-even
__device__ __forceinline__ bf16 f2bf(float x) {
    unsigned u = __float_as_uint(x);
    unsigned r = u + 0x7FFFu + ((u >> 16) & 1u);
    unsigned short h = (unsigned short)(r >> 16);
    return __builtin_bit_cast(bf16, h);
}

__global__ void cvt_f32_bf16(const float* __restrict__ in, bf16* __restrict__ out, int n) {
    int i = blockIdx.x * blockDim.x + threadIdx.x;
    int stride = gridDim.x * blockDim.x;
    for (; i < n; i += stride) out[i] = f2bf(in[i]);
}

__global__ void init_step(unsigned* minbits, float* alpha) {
    if (threadIdx.x == 0) { *minbits = 0xFFFFFFFFu; *alpha = 0.0f; }
}

// ---- wave-level GEMM core: one wave computes C[0:64, col0:col0+16] ----
// A: bf16 row-major (64 x K), W: bf16 row-major (N x K); C = A * W^T.
// 2x-unrolled ping-pong pipeline: compute buffer P while buffer 1-P loads.
// No rotation copies; K/32 is even for all call sites (256 / 64 / 32 chunks).
__device__ __forceinline__ void wave_gemm(const bf16* __restrict__ A,
                                          const bf16* __restrict__ W,
                                          int K, int col0, int lane, v8f acc[4]) {
    const int n = lane & 15;
    const int h = lane >> 4;
    const bf16* brow = W + (size_t)(col0 + n) * K;

    v16bf b0, b1;
    v8bf lo0[4], hi0[4], lo1[4], hi1[4];

    auto load_chunk = [&](int k0, v16bf& b, v8bf (&lo)[4], v8bf (&hi)[4]) {
        b = *(const v16bf*)(brow + k0 + h * 16);
#pragma unroll
        for (int q = 0; q < 4; ++q) {
            const bf16* ar = A + (size_t)(q * 16 + n) * K + k0;
            lo[q] = *(const v8bf*)(ar + 16 * h);
            hi[q] = *(const v8bf*)(ar + 16 * h + 8);
        }
    };
    auto compute = [&](const v16bf& b, const v8bf (&lo)[4], const v8bf (&hi)[4]) {
#pragma unroll
        for (int q = 0; q < 4; ++q) {
            v16bf afrag = __builtin_shufflevector(lo[q], hi[q],
                0, 1, 2, 3, 4, 5, 6, 7, 8, 9, 10, 11, 12, 13, 14, 15);
            acc[q] = __builtin_amdgcn_wmma_f32_16x16x32_bf16(
                false, afrag, false, b, (short)0, acc[q], false, false);
        }
    };

    load_chunk(0, b0, lo0, hi0);
    int k0 = 32;
    for (; k0 + 32 < K; k0 += 64) {
        load_chunk(k0, b1, lo1, hi1);
        if (k0 + 512 < K) __builtin_prefetch(brow + k0 + 512, 0, 1);
        compute(b0, lo0, hi0);
        load_chunk(k0 + 32, b0, lo0, hi0);
        compute(b1, lo1, hi1);
    }
    // K/32 even -> exactly one chunk left at k0 == K - 32
    load_chunk(k0, b1, lo1, hi1);
    compute(b0, lo0, hi0);
    compute(b1, lo1, hi1);
}

// generic GEMM + bias: C(64 x N) = A(64 x K) * W^T + bias
__global__ __launch_bounds__(256) void gemm_bias(const bf16* __restrict__ A,
                                                 const bf16* __restrict__ W,
                                                 const float* __restrict__ bias,
                                                 float* __restrict__ C, int K, int N) {
    int gw   = (blockIdx.x * blockDim.x + threadIdx.x) >> 5;
    int lane = threadIdx.x & 31;
    int col0 = gw * 16;
    if (col0 >= N) return;
    v8f acc[4];
    v8f zero = {0.f, 0.f, 0.f, 0.f, 0.f, 0.f, 0.f, 0.f};
    acc[0] = zero; acc[1] = zero; acc[2] = zero; acc[3] = zero;
    wave_gemm(A, W, K, col0, lane, acc);
    int n = lane & 15, h = lane >> 4;
    int col = col0 + n;
    float bv = bias[col];
#pragma unroll
    for (int q = 0; q < 4; ++q)
#pragma unroll
        for (int v = 0; v < 8; ++v) {
            int r = q * 16 + h * 8 + v;
            C[(size_t)r * N + col] = acc[q][v] + bv;
        }
}

// fused: sigma(64 x 8192) = x_b * Wb^T + bb + repeat(z_a, 8); track global min
__global__ __launch_bounds__(256) void gemm_zb_sigma(const bf16* __restrict__ xb,
                                                     const bf16* __restrict__ Wb,
                                                     const float* __restrict__ za,
                                                     const float* __restrict__ bb,
                                                     float* __restrict__ sigma,
                                                     unsigned* __restrict__ minbits) {
    int gw   = (blockIdx.x * blockDim.x + threadIdx.x) >> 5;
    int lane = threadIdx.x & 31;
    int col0 = gw * 16;
    if (col0 >= TCELL) return;
    v8f acc[4];
    v8f zero = {0.f, 0.f, 0.f, 0.f, 0.f, 0.f, 0.f, 0.f};
    acc[0] = zero; acc[1] = zero; acc[2] = zero; acc[3] = zero;
    wave_gemm(xb, Wb, TCELL, col0, lane, acc);
    int n = lane & 15, h = lane >> 4;
    int col = col0 + n;
    float bv = bb[col];
    int g = col >> 3;                 // group index for repeated z_a
    float lmin = 3.4e38f;
#pragma unroll
    for (int q = 0; q < 4; ++q)
#pragma unroll
        for (int v = 0; v < 8; ++v) {
            int r = q * 16 + h * 8 + v;
            float val = acc[q][v] + bv + za[(size_t)r * MDIM + g];
            sigma[(size_t)r * TCELL + col] = val;
            lmin = fminf(lmin, val);
        }
    atomicMin(minbits, orderbits(lmin));
}

// per-batch-row step: argmax within groups of 8, exact top-128 threshold via
// 32-round bitwise radix select, tanh, state updates, deterministic row sum
__global__ __launch_bounds__(256) void step_kernel(const float* __restrict__ sigma,
                                                   float* __restrict__ phi,
                                                   float* __restrict__ psi,
                                                   const unsigned* __restrict__ minbits,
                                                   bf16* __restrict__ gbf,
                                                   float* __restrict__ partial) {
    __shared__ int cnt;
    __shared__ float red[256];
    const int row = blockIdx.x;
    const int tid = threadIdx.x;
    const float minv = unorder(*minbits);

    unsigned b[4]; int am[4];
#pragma unroll
    for (int i = 0; i < 4; ++i) {
        int g = tid + i * 256;
        size_t base = (size_t)row * TCELL + (size_t)g * 8;
        float best = -3.4e38f; int bi = 0;
#pragma unroll
        for (int j = 0; j < 8; ++j) {
            float s  = sigma[base + j];
            float pi = (1.0f - phi[base + j]) * (s - minv + 1.0f);
            if (pi > best) { best = pi; bi = j; }
        }
        b[i] = orderbits(best); am[i] = bi;
    }
    __syncthreads();

    // kth-largest (K=128) of the 1024 lambda values for this row
    unsigned prefix = 0u;
    for (int bit = 31; bit >= 0; --bit) {
        unsigned T = prefix | (1u << bit);
        int c = (b[0] >= T) + (b[1] >= T) + (b[2] >= T) + (b[3] >= T);
        if (tid == 0) cnt = 0;
        __syncthreads();
        atomicAdd(&cnt, c);
        __syncthreads();
        if (cnt >= KSEL) prefix = T;
        __syncthreads();
    }

    float lsum = 0.0f;
#pragma unroll
    for (int i = 0; i < 4; ++i) {
        int g = tid + i * 256;
        size_t base = (size_t)row * TCELL + (size_t)g * 8;
        bool sel = (b[i] >= prefix);
        float ymax = 0.0f;
#pragma unroll
        for (int j = 0; j < 8; ++j) {
            float s  = sigma[base + j];
            float y  = (sel && j == am[i]) ? tanhf(s) : 0.0f;
            float ps = fmaxf(psi[base + j] * EPS_C,   y);
            float ph = fmaxf(phi[base + j] * GAMMA_C, y);
            psi[base + j] = ps;
            phi[base + j] = ph;
            lsum += ps;
            ymax = fmaxf(ymax, y);
        }
        gbf[(size_t)row * MDIM + g] = f2bf(ymax);   // bf16 feed for output GEMM
    }
    red[tid] = lsum;
    __syncthreads();
    for (int s = 128; s > 0; s >>= 1) {
        if (tid < s) red[tid] += red[tid + s];
        __syncthreads();
    }
    if (tid == 0) partial[row] = red[0];
}

__global__ void alpha_reduce(const float* __restrict__ partial, float* __restrict__ alpha) {
    __shared__ float red[64];
    int tid = threadIdx.x;
    red[tid] = partial[tid];
    __syncthreads();
    for (int s = 32; s > 0; s >>= 1) {
        if (tid < s) red[tid] += red[tid + s];
        __syncthreads();
    }
    if (tid == 0) { float a = red[0]; *alpha = (a == 0.0f) ? 1.0f : a; }
}

// x_b = psi_n / alpha ; emit fp32 state, bf16 GEMM feed, and x_bs[t]
__global__ void normalize_k(const float* __restrict__ psi, const float* __restrict__ alpha,
                            float* __restrict__ xb, bf16* __restrict__ xb_bf,
                            float* __restrict__ xbs_out) {
    int i = blockIdx.x * blockDim.x + threadIdx.x;
    if (i < BSZ * TCELL) {
        float v = psi[i] / *alpha;
        xb[i]      = v;
        xb_bf[i]   = f2bf(v);
        xbs_out[i] = v;
    }
}

extern "C" void kernel_launch(void* const* d_in, const int* in_sizes, int n_in,
                              void* d_out, int out_size, void* d_ws, size_t ws_size,
                              hipStream_t stream) {
    const float* x_a = (const float*)d_in[0];   // (64, 64, 2048)
    const float* xb0 = (const float*)d_in[1];   // (64, 8192)
    const float* ph0 = (const float*)d_in[2];
    const float* ps0 = (const float*)d_in[3];
    const float* Wa  = (const float*)d_in[4];   // (1024, 2048)
    const float* ba  = (const float*)d_in[5];
    const float* Wb  = (const float*)d_in[6];   // (8192, 8192)
    const float* bb  = (const float*)d_in[7];
    const float* Wd  = (const float*)d_in[8];   // (2048, 1024)
    const float* bd  = (const float*)d_in[9];

    char* ws = (char*)d_ws;
    bf16* Wb_bf = (bf16*)(ws + WS_WB);
    bf16* Wa_bf = (bf16*)(ws + WS_WA);
    bf16* Wd_bf = (bf16*)(ws + WS_WD);
    bf16* xa_bf = (bf16*)(ws + WS_XA_BF);
    bf16* xb_bf = (bf16*)(ws + WS_XBBF);
    bf16* g_bf  = (bf16*)(ws + WS_GBF);
    float* xb_s  = (float*)(ws + WS_XB);
    float* phi_s = (float*)(ws + WS_PHI);
    float* psi_s = (float*)(ws + WS_PSI);
    float* za    = (float*)(ws + WS_ZA);
    float* sig   = (float*)(ws + WS_SIG);
    float* part  = (float*)(ws + WS_PART);
    unsigned* minbits = (unsigned*)(ws + WS_SCAL);
    float*    alpha   = (float*)(ws + WS_SCAL + 4);
    float* out = (float*)d_out;

    // one-time (per call) conversions: weights + whole x_a batch + x_b0
    cvt_f32_bf16<<<4096, 256, 0, stream>>>(Wb, Wb_bf, TCELL * TCELL);
    cvt_f32_bf16<<<512,  256, 0, stream>>>(Wa, Wa_bf, MDIM * D_IN);
    cvt_f32_bf16<<<512,  256, 0, stream>>>(Wd, Wd_bf, D_OUT * MDIM);
    cvt_f32_bf16<<<2048, 256, 0, stream>>>(x_a, xa_bf, SEQ * BSZ * D_IN);
    cvt_f32_bf16<<<512,  256, 0, stream>>>(xb0, xb_bf, BSZ * TCELL);

    // init recurrent state
    hipMemcpyAsync(xb_s,  xb0, (size_t)BSZ * TCELL * 4, hipMemcpyDeviceToDevice, stream);
    hipMemcpyAsync(phi_s, ph0, (size_t)BSZ * TCELL * 4, hipMemcpyDeviceToDevice, stream);
    hipMemcpyAsync(psi_s, ps0, (size_t)BSZ * TCELL * 4, hipMemcpyDeviceToDevice, stream);

    for (int t = 0; t < SEQ; ++t) {
        init_step<<<1, 32, 0, stream>>>(minbits, alpha);
        // z_a = x_a[t] @ Wa^T + ba : 64 col-tiles -> 64 waves
        gemm_bias<<<8, 256, 0, stream>>>(xa_bf + (size_t)t * BSZ * D_IN, Wa_bf, ba, za,
                                         D_IN, MDIM);
        // sigma = x_b @ Wb^T + bb + repeat(z_a) ; global min : 512 waves
        gemm_zb_sigma<<<64, 256, 0, stream>>>(xb_bf, Wb_bf, za, bb, sig, minbits);
        // masks, tanh, state update, group-max (bf16), row sums
        step_kernel<<<BSZ, 256, 0, stream>>>(sig, phi_s, psi_s, minbits, g_bf, part);
        alpha_reduce<<<1, 64, 0, stream>>>(part, alpha);
        // x_b = psi_n / alpha ; emit x_bs[t] and bf16 x_b
        normalize_k<<<(BSZ * TCELL + 255) / 256, 256, 0, stream>>>(
            psi_s, alpha, xb_s, xb_bf, out + O_XBS + (size_t)t * BSZ * TCELL);
        // output[t] = group_max(y) @ Wd^T + bd : 128 waves
        gemm_bias<<<16, 256, 0, stream>>>(g_bf, Wd_bf, bd,
                                          out + O_OUT + (size_t)t * BSZ * D_OUT, MDIM, D_OUT);
    }

    // final carry outputs
    hipMemcpyAsync(out + O_XB,  xb_s,  (size_t)BSZ * TCELL * 4, hipMemcpyDeviceToDevice, stream);
    hipMemcpyAsync(out + O_PHI, phi_s, (size_t)BSZ * TCELL * 4, hipMemcpyDeviceToDevice, stream);
    hipMemcpyAsync(out + O_PSI, psi_s, (size_t)BSZ * TCELL * 4, hipMemcpyDeviceToDevice, stream);
}
